// AttnEncoder_73065983639736
// MI455X (gfx1250) — compile-verified, matrix operand order checked by reading, not run on hardware
//
#include <hip/hip_runtime.h>
#include <hip/hip_bf16.h>

// ---------------- types ----------------
typedef __bf16          bf16x16 __attribute__((ext_vector_type(16)));
typedef float           f32x8   __attribute__((ext_vector_type(8)));
typedef unsigned short  u16x8   __attribute__((ext_vector_type(8)));

struct AFragBits { u16x8 lo, hi; };   // 32 bytes -> bit_cast to bf16x16

// ---------------- sizes ----------------
#define B_SZ  1024
#define T_SZ  128
#define DPV   10
#define HID   512
#define INP   512
#define NG    2048       // 4*HID gate width
#define KT    32         // K tiles (1024 / 32)
#define NT1   32         // N tiles for z1/x GEMMs (512/16)
#define NTG   128        // N tiles for gate GEMM (2048/16)
#define FRAG  512        // ushorts per packed B fragment (32 lanes x 16)

__device__ __forceinline__ unsigned short f2bf(float f) {
    unsigned u = __builtin_bit_cast(unsigned, f);
    u += 0x7FFFu + ((u >> 16) & 1u);          // round-to-nearest-even
    return (unsigned short)(u >> 16);
}
__device__ __forceinline__ unsigned pack2bf(float lo, float hi) {
    return (unsigned)f2bf(lo) | ((unsigned)f2bf(hi) << 16);
}

// CDNA5 16-bit WMMA operand K mapping (ISA 7.12.2):
// VGPR v holds K pair; lanes>=16 shift +8; v>=4 covers K 16..31.
__device__ __forceinline__ int kmap(int lane, int e) {
    int v = e >> 1;
    return ((v >= 4) ? 16 : 0) + ((lane >= 16) ? 8 : 0) + ((v & 3) << 1) + (e & 1);
}

__device__ __forceinline__ f32x8 zero8() {
    f32x8 z;
#pragma unroll
    for (int i = 0; i < 8; ++i) z[i] = 0.f;
    return z;
}

__device__ __forceinline__ float sigm(float x) { return 1.f / (1.f + __expf(-x)); }

// Ping-pong pipelined K-loop GEMM (kt advances by 2): even iterations consume
// bufE while bufO's global_load_b128s are in flight, and vice versa. No
// register rotation copies -> no cndmask moves in the inner loop.
// A rows come from (a0,s0) for kt<ktSplit, else (a1,s1)  (lets gates read [x|h]).
template<int NJ>
__device__ __forceinline__ void gemm_pipelined(
    const unsigned short* __restrict__ a0, int s0, int ktSplit,
    const unsigned short* __restrict__ a1, int s1,
    const unsigned short* __restrict__ wfrag0,   // frag(kt=0, ntbase) + lane*16
    size_t ktStride,                             // ushorts per kt step (NTILES*FRAG)
    int mA, int koff,
    f32x8* acc)
{
    auto loadB = [&](int kt, bf16x16* dst) {
        const unsigned short* wp = wfrag0 + (size_t)kt * ktStride;
#pragma unroll
        for (int j = 0; j < NJ; ++j)
            dst[j] = *(const bf16x16*)(wp + (size_t)j * FRAG);
    };
    auto loadA = [&](int kt) -> bf16x16 {
        const unsigned short* arow = (kt < ktSplit)
            ? (a0 + mA * s0 + kt * 32 + koff)
            : (a1 + mA * s1 + (kt - ktSplit) * 32 + koff);
        AFragBits ab;
        ab.lo = *(const u16x8*)(arow);
        ab.hi = *(const u16x8*)(arow + 16);
        return __builtin_bit_cast(bf16x16, ab);
    };

    __builtin_prefetch(wfrag0 + ktStride, 0, 2);
    bf16x16 bufE[NJ], bufO[NJ];
    loadB(0, bufE);
#pragma unroll 1
    for (int kt = 0; kt < KT; kt += 2) {
        loadB(kt + 1, bufO);                 // in flight during even WMMAs
        bf16x16 afE = loadA(kt);
        bf16x16 afO = loadA(kt + 1);
#pragma unroll
        for (int j = 0; j < NJ; ++j)
            acc[j] = __builtin_amdgcn_wmma_f32_16x16x32_bf16(
                false, afE, false, bufE[j], (short)0, acc[j], false, false);
        if (kt + 2 < KT) loadB(kt + 2, bufE); // in flight during odd WMMAs
#pragma unroll
        for (int j = 0; j < NJ; ++j)
            acc[j] = __builtin_amdgcn_wmma_f32_16x16x32_bf16(
                false, afO, false, bufO[j], (short)0, acc[j], false, false);
    }
}

// ---------------- weight prepack: (N,K) f32 row-major -> bf16 WMMA B fragments ----------------
// k < ksplit reads pa[n*sa+k], else pb[n*sb+(k-ksplit)]  (fuses [Wih|Whh] along K)
__global__ void pack_w_kernel(const float* __restrict__ pa, int sa,
                              const float* __restrict__ pb, int sb, int ksplit,
                              int ntiles, unsigned short* __restrict__ dst) {
    int idx = blockIdx.x * blockDim.x + threadIdx.x;
    int total = KT * ntiles * FRAG;
    if (idx >= total) return;
    int frag   = idx / FRAG;
    int within = idx % FRAG;
    int lane = within >> 4, e = within & 15;
    int kt = frag / ntiles, nt = frag % ntiles;
    int n = nt * 16 + (lane & 15);
    int k = kt * 32 + kmap(lane, e);
    float v = (k < ksplit) ? pa[(size_t)n * sa + k] : pb[(size_t)n * sb + (k - ksplit)];
    dst[idx] = f2bf(v);
}

__global__ void pack_bias_kernel(const float* __restrict__ bih,
                                 const float* __restrict__ bhh,
                                 float* __restrict__ bg) {
    int i = blockIdx.x * blockDim.x + threadIdx.x;
    if (i < NG) bg[i] = bih[i] + bhh[i];
}

// ---------------- persistent per-batch-tile LSTM ----------------
// grid = 64 blocks (16 batch rows each), block = 256 threads = 8 wave32
__global__ __launch_bounds__(256, 1)
void lstm_persistent_kernel(const float* __restrict__ sv,   // [B,T,10]
                            const float* __restrict__ b1,
                            const float* __restrict__ W2,   // [512,10]
                            const float* __restrict__ b2,
                            const float* __restrict__ b3,
                            const unsigned short* __restrict__ w1p,
                            const unsigned short* __restrict__ w3p,
                            const unsigned short* __restrict__ wgp,
                            const float* __restrict__ bg,
                            float* __restrict__ out)        // [B,T,512]
{
    extern __shared__ char smem[];
    unsigned short* bufA = (unsigned short*)smem;          // 16x1024 bf16: [h | c]
    unsigned short* bufB = bufA + 16 * 1024;               // 16x1024 bf16: [z1 | z2]
    unsigned short* bufX = bufB + 16 * 1024;               // 16x512  bf16: x
    float* gatesS = (float*)(bufX + 16 * 512);             // 16x2048 f32
    float* W2t    = gatesS + 16 * 2048;                    // [10][512] f32 (transposed)
    float* b1s    = W2t + DPV * INP;
    float* b2s    = b1s + INP;
    float* b3s    = b2s + INP;
    float* bgs    = b3s + INP;                             // 2048 f32

    const int tid  = threadIdx.x;
    const int lane = tid & 31;
    const int wv   = tid >> 5;
    const int b0   = blockIdx.x * 16;

    // one-time staging: transposed W2 + all biases into LDS
    for (int i = tid; i < DPV * INP; i += 256) {
        int d = i / INP, j = i % INP;
        W2t[i] = W2[(size_t)j * DPV + d];
    }
    for (int i = tid; i < INP; i += 256) {
        b1s[i] = b1[i]; b2s[i] = b2[i]; b3s[i] = b3[i];
    }
    for (int i = tid; i < NG; i += 256) bgs[i] = bg[i];
    for (int i = tid; i < 16 * 1024; i += 256) bufA[i] = 0;   // h0 = c0 = 0

    // elementwise ownership: 16 threads per batch row, 32 hidden units each.
    const int erow = tid >> 4;
    const int ej0  = (tid & 15) * 32;
    float creg[32];
#pragma unroll
    for (int i = 0; i < 32; ++i) creg[i] = 0.f;

    __syncthreads();

    const int mA   = lane & 15;               // A-frag row / C-tile column
    const int koff = (lane >= 16) ? 8 : 0;

    for (int t = 0; t < T_SZ; ++t) {
        // ===== phase A: z2 (VALU, LDS-resident W2t) + z1 GEMM [h|c] x W1^T =====
        {
            float sv10[DPV];
            const float* svp = sv + ((size_t)(b0 + erow) * T_SZ + t) * DPV;
#pragma unroll
            for (int d = 0; d < DPV; ++d) sv10[d] = svp[d];
            float a32[32];
#pragma unroll
            for (int jj = 0; jj < 32; ++jj) a32[jj] = b2s[ej0 + jj];
#pragma unroll
            for (int d = 0; d < DPV; ++d) {
                float s = sv10[d];
#pragma unroll
                for (int jj = 0; jj < 32; ++jj)
                    a32[jj] += s * W2t[d * INP + ej0 + jj];
            }
            unsigned* z2p = (unsigned*)(bufB + erow * 1024 + 512 + ej0);
#pragma unroll
            for (int jj = 0; jj < 32; jj += 2) {
                float alo = a32[jj], ahi = a32[jj + 1];
                z2p[jj >> 1] = pack2bf(alo > 0.f ? alo : 0.f, ahi > 0.f ? ahi : 0.f);
            }

            f32x8 acc[4];
#pragma unroll
            for (int j = 0; j < 4; ++j) acc[j] = zero8();
            gemm_pipelined<4>(bufA, 1024, KT, bufA, 1024,
                              w1p + (size_t)(wv * 4) * FRAG + lane * 16,
                              (size_t)NT1 * FRAG, mA, koff, acc);
#pragma unroll
            for (int j = 0; j < 4; ++j) {
                int ncol = wv * 64 + j * 16 + mA;
                float bias = b1s[ncol];
#pragma unroll
                for (int r = 0; r < 8; ++r) {
                    int m = (lane < 16) ? r : (r + 8);
                    float v = acc[j][r] + bias;
                    bufB[m * 1024 + ncol] = f2bf(v > 0.f ? v : 0.f);
                }
            }
        }
        __syncthreads();

        // ===== phase B: x GEMM [z1|z2] x W3^T =====
        {
            f32x8 acc[4];
#pragma unroll
            for (int j = 0; j < 4; ++j) acc[j] = zero8();
            gemm_pipelined<4>(bufB, 1024, KT, bufB, 1024,
                              w3p + (size_t)(wv * 4) * FRAG + lane * 16,
                              (size_t)NT1 * FRAG, mA, koff, acc);
#pragma unroll
            for (int j = 0; j < 4; ++j) {
                int ncol = wv * 64 + j * 16 + mA;
                float bias = b3s[ncol];
#pragma unroll
                for (int r = 0; r < 8; ++r) {
                    int m = (lane < 16) ? r : (r + 8);
                    float v = acc[j][r] + bias;
                    bufX[m * 512 + ncol] = f2bf(v > 0.f ? v : 0.f);
                }
            }
        }
        __syncthreads();

        // ===== phase C: gate GEMM [x|h] x [Wih|Whh]^T, two halves of 8 N-tiles =====
#pragma unroll 1
        for (int hh = 0; hh < 2; ++hh) {
            f32x8 acc[8];
#pragma unroll
            for (int j = 0; j < 8; ++j) acc[j] = zero8();
            gemm_pipelined<8>(bufX, 512, 16, bufA, 1024,      // kt<16: x, kt>=16: h
                              wgp + (size_t)(wv * 16 + hh * 8) * FRAG + lane * 16,
                              (size_t)NTG * FRAG, mA, koff, acc);
#pragma unroll
            for (int j = 0; j < 8; ++j) {
                int ncol = wv * 256 + hh * 128 + j * 16 + mA;
                float bias = bgs[ncol];
#pragma unroll
                for (int r = 0; r < 8; ++r) {
                    int m = (lane < 16) ? r : (r + 8);
                    gatesS[m * 2048 + ncol] = acc[j][r] + bias;
                }
            }
        }
        __syncthreads();

        // ===== phase D: elementwise LSTM; c persists in registers =====
        {
            float* outp = out + ((size_t)(b0 + erow) * T_SZ + t) * HID;
            const float* g = gatesS + erow * 2048;
            unsigned* hp = (unsigned*)(bufA + erow * 1024 + ej0);         // h half
            unsigned* cp = (unsigned*)(bufA + erow * 1024 + 512 + ej0);   // c half
#pragma unroll 4
            for (int jj = 0; jj < 32; jj += 2) {
                int j = ej0 + jj;
                float h2[2], c2[2];
#pragma unroll
                for (int q = 0; q < 2; ++q) {
                    float ig = sigm(g[j + q]);
                    float fg = sigm(g[512 + j + q]);
                    float gg = tanhf(g[1024 + j + q]);
                    float og = sigm(g[1536 + j + q]);
                    float c  = fg * creg[jj + q] + ig * gg;
                    creg[jj + q] = c;
                    c2[q] = c;
                    h2[q] = og * tanhf(c);
                    outp[j + q] = h2[q];
                }
                hp[jj >> 1] = pack2bf(h2[0], h2[1]);
                cp[jj >> 1] = pack2bf(c2[0], c2[1]);
            }
        }
        __syncthreads();
    }
}

extern "C" void kernel_launch(void* const* d_in, const int* in_sizes, int n_in,
                              void* d_out, int out_size, void* d_ws, size_t ws_size,
                              hipStream_t stream) {
    (void)in_sizes; (void)n_in; (void)out_size; (void)ws_size;
    const float* sv  = (const float*)d_in[0];
    const float* W1  = (const float*)d_in[1];
    const float* b1  = (const float*)d_in[2];
    const float* W2  = (const float*)d_in[3];
    const float* b2  = (const float*)d_in[4];
    const float* W3  = (const float*)d_in[5];
    const float* b3  = (const float*)d_in[6];
    const float* Wih = (const float*)d_in[7];
    const float* Whh = (const float*)d_in[8];
    const float* bih = (const float*)d_in[9];
    const float* bhh = (const float*)d_in[10];
    float* out = (float*)d_out;

    // workspace: packed bf16 weights + fused gate bias (~6 MB)
    unsigned short* w1p = (unsigned short*)d_ws;
    unsigned short* w3p = w1p + (size_t)KT * NT1 * FRAG;
    unsigned short* wgp = w3p + (size_t)KT * NT1 * FRAG;
    float*          bgp = (float*)(wgp + (size_t)KT * NTG * FRAG);

    const int tw1 = KT * NT1 * FRAG;     // 524288
    const int twg = KT * NTG * FRAG;     // 2097152
    pack_w_kernel<<<(tw1 + 255) / 256, 256, 0, stream>>>(W1, 1024, W1, 1024, 1024, NT1, w1p);
    pack_w_kernel<<<(tw1 + 255) / 256, 256, 0, stream>>>(W3, 1024, W3, 1024, 1024, NT1, w3p);
    pack_w_kernel<<<(twg + 255) / 256, 256, 0, stream>>>(Wih, 512, Whh, 512, 512, NTG, wgp);
    pack_bias_kernel<<<(NG + 255) / 256, 256, 0, stream>>>(bih, bhh, bgp);

    const size_t smem =
        (size_t)(16 * 1024 + 16 * 1024 + 16 * 512) * sizeof(unsigned short)       // bufA/B/X
        + (size_t)(16 * 2048 + DPV * INP + 3 * INP + NG) * sizeof(float);         // gates+W2t+biases
    lstm_persistent_kernel<<<B_SZ / 16, 256, smem, stream>>>(
        sv, b1, W2, b2, b3, w1p, w3p, wgp, bgp, out);
}
